// CrossAttentionHead_46325517254778
// MI455X (gfx1250) — matrix operand order repaired and checked by reading, not compiled, not run on hardware
//
#include <hip/hip_runtime.h>
#include <hip/hip_fp16.h>

typedef __attribute__((ext_vector_type(16))) _Float16     v16h;
typedef __attribute__((ext_vector_type(8)))  _Float16     v8h;
typedef __attribute__((ext_vector_type(8)))  float        v8f;
typedef __attribute__((ext_vector_type(4)))  unsigned int u32x4;
typedef __attribute__((ext_vector_type(8)))  int          i32x8;
typedef __attribute__((ext_vector_type(4)))  int          i32x4;

#define D_MODEL 1024
#define QKV     64
#define T_LEN   2048
#define S_LEN   2048
#define BATCH   8

// ---------------------------------------------------------------------------
// TDM helper: 2D tile load (f16 elements) global -> LDS via the Tensor Data
// Mover (D# per CDNA5 ISA).  Groups 2/3 zero (2D tile,
// tile_dim2=0).  6-arg builtin variant (clang-23 lane).  Tracked w/ TENSORcnt.
// ---------------------------------------------------------------------------
__device__ __forceinline__ void tdm_load_2d(unsigned lds_addr,
                                            const _Float16* gptr,
                                            unsigned tensor_d0, unsigned tensor_d1,
                                            unsigned long stride0,   // elems
                                            unsigned tile_d0, unsigned tile_d1)
{
    const unsigned long ga = (unsigned long)gptr;
    u32x4 g0;
    g0[0] = 1u;                                        // count=1, user mode
    g0[1] = lds_addr;                                  // LDS byte address
    g0[2] = (unsigned)(ga & 0xffffffffu);              // global_addr[31:0]
    g0[3] = (unsigned)((ga >> 32) & 0x01ffffffu)       // global_addr[56:32]
          | (2u << 30);                                // type=2 ("image")
    i32x8 g1;
    g1[0] = (1 << 16);                                 // wg_mask=0, data_size=1 (2B)
    g1[1] = (int)(tensor_d0 << 16);                    // tensor_dim0[15:0] @63:48
    g1[2] = (int)((tensor_d0 >> 16) | (tensor_d1 << 16));
    g1[3] = (int)((tensor_d1 >> 16) | (tile_d0 << 16));
    g1[4] = (int)tile_d1;                              // tile_dim1; tile_dim2=0
    g1[5] = (int)(stride0 & 0xffffffffu);              // dim0 stride [47:0]
    g1[6] = (int)((stride0 >> 32) & 0xffffu);
    g1[7] = 0;
    const i32x4 z4 = {0, 0, 0, 0};
    const i32x8 z8 = {0, 0, 0, 0, 0, 0, 0, 0};
    __builtin_amdgcn_tensor_load_to_lds(g0, g1, z4, z4, z8, 0);
}

// ---------------------------------------------------------------------------
// Kernel 0: weight prep.  Wt[path][n][k] = (f16) W[path][k][n].
// ---------------------------------------------------------------------------
__global__ __launch_bounds__(256)
void prep_w_kernel(const float* __restrict__ Wq,
                   const float* __restrict__ Wk,
                   const float* __restrict__ Wv,
                   _Float16* __restrict__ Wt)
{
    const int idx = blockIdx.x * 256 + threadIdx.x;   // 0 .. 3*64*1024-1
    const int p   = idx >> 16;
    const int rem = idx & 65535;
    const int n   = rem >> 10;
    const int k   = rem & 1023;
    const float* W = (p == 0) ? Wq : (p == 1) ? Wk : Wv;
    Wt[idx] = (_Float16)W[k * QKV + n];
}

// ---------------------------------------------------------------------------
// Kernel 1: fused QKV projection (fp32 in -> f16 out).  b128 A loads + pack,
// v16h B loads from transposed f16 weights, V stored transposed Vt[b][d][s].
// ---------------------------------------------------------------------------
__global__ __launch_bounds__(32)
void qkv_proj_kernel(const float* __restrict__ query,
                     const float* __restrict__ key_value,
                     const _Float16* __restrict__ Wt,
                     const float* __restrict__ bq,
                     const float* __restrict__ bk,
                     const float* __restrict__ bv,
                     _Float16* __restrict__ Qb,
                     _Float16* __restrict__ Kb,
                     _Float16* __restrict__ Vt)
{
    const int lane   = threadIdx.x;
    const int half   = lane >> 4;
    const int lane16 = lane & 15;
    const int tile   = blockIdx.x;
    const int path   = blockIdx.y;

    const float*    in   = (path == 0) ? query : key_value;
    const _Float16* WtP  = Wt + (long)path * QKV * D_MODEL;   // [n][k]
    const float*    bias = (path == 0) ? bq : (path == 1) ? bk : bv;

    v8f acc[4] = {};
    const long rowA = (long)tile * 16 + lane16;
    const float* arow = in + rowA * D_MODEL;

    for (int k0 = 0; k0 < D_MODEL; k0 += 32) {
        const float* ap = arow + k0 + half * 8;
        const float4 q0 = *(const float4*)(ap);
        const float4 q1 = *(const float4*)(ap + 4);
        const float4 q2 = *(const float4*)(ap + 16);
        const float4 q3 = *(const float4*)(ap + 20);
        v16h a;
        a[0]=(_Float16)q0.x; a[1]=(_Float16)q0.y; a[2]=(_Float16)q0.z; a[3]=(_Float16)q0.w;
        a[4]=(_Float16)q1.x; a[5]=(_Float16)q1.y; a[6]=(_Float16)q1.z; a[7]=(_Float16)q1.w;
        a[8]=(_Float16)q2.x; a[9]=(_Float16)q2.y; a[10]=(_Float16)q2.z; a[11]=(_Float16)q2.w;
        a[12]=(_Float16)q3.x; a[13]=(_Float16)q3.y; a[14]=(_Float16)q3.z; a[15]=(_Float16)q3.w;

#pragma unroll
        for (int nt = 0; nt < 4; ++nt) {
            const int n = nt * 16 + lane16;
            const _Float16* wp = WtP + (long)n * D_MODEL + k0 + half * 8;
            const v16h bm = __builtin_shufflevector(*(const v8h*)(wp), *(const v8h*)(wp + 16), 0,1,2,3,4,5,6,7,8,9,10,11,12,13,14,15);
            acc[nt] = __builtin_amdgcn_wmma_f32_16x16x32_f16(
                false, a, false, bm, (short)0, acc[nt], false, false);
        }
    }

#pragma unroll
    for (int nt = 0; nt < 4; ++nt) {
        const int n  = nt * 16 + lane16;
        const float bb = bias[n];
#pragma unroll
        for (int r = 0; r < 8; ++r) {
            const long row = (long)tile * 16 + r + half * 8;
            const _Float16 v = (_Float16)(acc[nt][r] + bb);
            if (path == 0) {
                Qb[row * QKV + n] = v;
            } else if (path == 1) {
                Kb[row * QKV + n] = v;
            } else {
                const long bidx = row >> 11;
                const long t    = row & 2047;
                Vt[(bidx * QKV + n) * S_LEN + t] = v;
            }
        }
    }
}

// ---------------------------------------------------------------------------
// Kernel 2: flash attention with TDM double buffering.
// One wave per (batch, 16 query rows).  K (64x64) and Vt (64x64) tiles are
// DMA'd to LDS by tensor_load_to_lds one chunk ahead (TENSORcnt), WMMA B
// operands are contiguous ds_load_b128 runs, probs bounce through LDS.
// ---------------------------------------------------------------------------
__global__ __launch_bounds__(32)
void attention_kernel(const _Float16* __restrict__ Qb,
                      const _Float16* __restrict__ Kb,
                      const _Float16* __restrict__ Vt,
                      const int* __restrict__ mask,
                      float* __restrict__ out)
{
    __shared__ _Float16 ktile[2][64 * 64];   // [s_local][d], 8 KB per buffer
    __shared__ _Float16 vtile[2][64 * 64];   // [d][s_local], 8 KB per buffer
    __shared__ _Float16 pshare[16 * 64];     // probs staging, 2 KB

    const int lane   = threadIdx.x;
    const int half   = lane >> 4;
    const int lane16 = lane & 15;
    const int t0     = blockIdx.x * 16;
    const int b      = blockIdx.y;

    const _Float16* Qbase = Qb + ((long)b * T_LEN + t0) * QKV;
    const _Float16* Kbase = Kb + (long)b * S_LEN * QKV;
    const _Float16* Vtb   = Vt + (long)b * QKV * S_LEN;     // [d][s]
    const int*      mrow  = mask + (long)b * S_LEN;

    const unsigned lds_k0 = (unsigned)(size_t)&ktile[0][0];
    const unsigned lds_k1 = (unsigned)(size_t)&ktile[1][0];
    const unsigned lds_v0 = (unsigned)(size_t)&vtile[0][0];
    const unsigned lds_v1 = (unsigned)(size_t)&vtile[1][0];

    // Q tile as two A operands (two contiguous 16B runs per lane)
    v16h aq[2];
#pragma unroll
    for (int kc = 0; kc < 2; ++kc) {
        const _Float16* qp = Qbase + (long)lane16 * QKV + kc * 32 + half * 8;
        const v8h lo = *(const v8h*)(qp);
        const v8h hi = *(const v8h*)(qp + 16);
        aq[kc] = __builtin_shufflevector(lo, hi, 0,1,2,3,4,5,6,7,
                                                 8,9,10,11,12,13,14,15);
    }

    float row_max[8], row_sum[8];
    v8f   oacc[4] = {};
#pragma unroll
    for (int r = 0; r < 8; ++r) { row_max[r] = -1e30f; row_sum[r] = 0.0f; }

    // prologue: DMA chunk 0 into buffer 0
    tdm_load_2d(lds_k0, Kbase, QKV, S_LEN, QKV, 64, 64);
    tdm_load_2d(lds_v0, Vtb,  S_LEN, QKV, S_LEN, 64, 64);

    for (int ci = 0; ci < S_LEN / 64; ++ci) {
        const int s0  = ci * 64;
        const int buf = ci & 1;
        const _Float16* Kls = buf ? &ktile[1][0] : &ktile[0][0];
        const _Float16* Vls = buf ? &vtile[1][0] : &vtile[0][0];

        if (ci + 1 < S_LEN / 64) {
            // previous reads of the other buffer must have retired
            asm volatile("s_wait_dscnt 0" ::: "memory");
            const unsigned nk = buf ? lds_k0 : lds_k1;
            const unsigned nv = buf ? lds_v0 : lds_v1;
            tdm_load_2d(nk, Kbase + (long)(s0 + 64) * QKV, QKV, S_LEN, QKV, 64, 64);
            tdm_load_2d(nv, Vtb + (s0 + 64),  S_LEN, QKV, S_LEN, 64, 64);
            __builtin_amdgcn_s_wait_tensorcnt(2);   // this chunk's 2 DMAs done
        } else {
            __builtin_amdgcn_s_wait_tensorcnt(0);
        }

        // ---- scores: 16x64 chunk = Q * K^T ; B from LDS (contig. in d)
        v8f sc[4] = {};
#pragma unroll
        for (int nt = 0; nt < 4; ++nt) {
            const int sl = nt * 16 + lane16;        // column N = key row
#pragma unroll
            for (int kc = 0; kc < 2; ++kc) {
                const _Float16* kp = Kls + sl * 64 + kc * 32 + half * 8;
                const v16h bk_ = __builtin_shufflevector(*(const v8h*)(kp), *(const v8h*)(kp + 16), 0,1,2,3,4,5,6,7,8,9,10,11,12,13,14,15);
                sc[nt] = __builtin_amdgcn_wmma_f32_16x16x32_f16(
                    false, aq[kc], false, bk_, (short)0, sc[nt], false, false);
            }
        }

        // ---- scale + mask
#pragma unroll
        for (int nt = 0; nt < 4; ++nt) {
            const int  s     = s0 + nt * 16 + lane16;
            const bool valid = (mrow[s] != 0);
#pragma unroll
            for (int r = 0; r < 8; ++r) {
                const float v = sc[nt][r] * 0.125f;   // 1/sqrt(64)
                sc[nt][r] = valid ? v : -2e30f;
            }
        }

        // ---- per-row chunk max: fold tiles in-lane, 16-lane butterfly
        float cmax[8];
#pragma unroll
        for (int r = 0; r < 8; ++r) {
            float m = fmaxf(fmaxf(sc[0][r], sc[1][r]), fmaxf(sc[2][r], sc[3][r]));
#pragma unroll
            for (int off = 1; off < 16; off <<= 1)
                m = fmaxf(m, __shfl_xor(m, off, 32));
            cmax[r] = m;
        }

        // ---- online softmax rescale
#pragma unroll
        for (int r = 0; r < 8; ++r) {
            const float nm    = fmaxf(row_max[r], cmax[r]);
            const float alpha = __expf(row_max[r] - nm);
            row_max[r]  = nm;
            row_sum[r] *= alpha;
#pragma unroll
            for (int dt = 0; dt < 4; ++dt)
                oacc[dt][r] *= alpha;
        }

        // ---- p = exp(score - max): row sums + stage P in LDS
#pragma unroll
        for (int r = 0; r < 8; ++r) {
            const int prow = r + half * 8;
            float psum = 0.0f;
#pragma unroll
            for (int nt = 0; nt < 4; ++nt) {
                const float p = __expf(sc[nt][r] - row_max[r]);
                psum += p;
                pshare[prow * 64 + nt * 16 + lane16] = (_Float16)p;
            }
#pragma unroll
            for (int off = 1; off < 16; off <<= 1)
                psum += __shfl_xor(psum, off, 32);
            row_sum[r] += psum;
        }

        asm volatile("s_wait_dscnt 0" ::: "memory");

        // ---- reload P as A operands (two ds_load_b128 per operand)
        v16h ap[2];
#pragma unroll
        for (int kc = 0; kc < 2; ++kc) {
            const _Float16* pp = pshare + lane16 * 64 + kc * 32 + half * 8;
            const v8h lo = *(const v8h*)(pp);
            const v8h hi = *(const v8h*)(pp + 16);
            ap[kc] = __builtin_shufflevector(lo, hi, 0,1,2,3,4,5,6,7,
                                                     8,9,10,11,12,13,14,15);
        }

        // ---- O += P * V ; B from LDS Vt tile (contiguous in s)
#pragma unroll
        for (int dt = 0; dt < 4; ++dt) {
            const int d = dt * 16 + lane16;
#pragma unroll
            for (int kc = 0; kc < 2; ++kc) {
                const _Float16* vp = Vls + d * 64 + kc * 32 + half * 8;
                const v16h bv_ = __builtin_shufflevector(*(const v8h*)(vp), *(const v8h*)(vp + 16), 0,1,2,3,4,5,6,7,8,9,10,11,12,13,14,15);
                oacc[dt] = __builtin_amdgcn_wmma_f32_16x16x32_f16(
                    false, ap[kc], false, bv_, (short)0, oacc[dt], false, false);
            }
        }
    }

    // ---- normalize and store fp32 output
#pragma unroll
    for (int r = 0; r < 8; ++r) {
        const float inv  = 1.0f / row_sum[r];
        const long  trow = (long)t0 + r + half * 8;
#pragma unroll
        for (int dt = 0; dt < 4; ++dt) {
            const int d = dt * 16 + lane16;
            out[((long)b * T_LEN + trow) * QKV + d] = oacc[dt][r] * inv;
        }
    }
}

// ---------------------------------------------------------------------------
extern "C" void kernel_launch(void* const* d_in, const int* in_sizes, int n_in,
                              void* d_out, int out_size, void* d_ws, size_t ws_size,
                              hipStream_t stream) {
    const float* query     = (const float*)d_in[0];
    const float* key_value = (const float*)d_in[1];
    const int*   mask      = (const int*)  d_in[2];
    const float* Wq        = (const float*)d_in[3];
    const float* bq        = (const float*)d_in[4];
    const float* Wk        = (const float*)d_in[5];
    const float* bk        = (const float*)d_in[6];
    const float* Wv        = (const float*)d_in[7];
    const float* bv        = (const float*)d_in[8];
    float*       out       = (float*)d_out;

    const size_t elems = (size_t)BATCH * T_LEN * QKV;   // 1M halves each
    _Float16* Qb = (_Float16*)d_ws;
    _Float16* Kb = Qb + elems;
    _Float16* Vt = Kb + elems;
    _Float16* Wt = Vt + elems;                          // 3*64*1024 halves

    prep_w_kernel<<<(3 * QKV * D_MODEL) / 256, 256, 0, stream>>>(Wq, Wk, Wv, Wt);

    dim3 g1((BATCH * T_LEN) / 16, 3);
    qkv_proj_kernel<<<g1, 32, 0, stream>>>(query, key_value, Wt,
                                           bq, bk, bv, Qb, Kb, Vt);

    dim3 g2(T_LEN / 16, BATCH);
    attention_kernel<<<g2, 32, 0, stream>>>(Qb, Kb, Vt, mask, out);
}
